// BlockLocalAttention_56925496541795
// MI455X (gfx1250) — matrix operand order repaired and checked by reading, not compiled, not run on hardware
//
#include <hip/hip_runtime.h>
#include <hip/hip_bf16.h>

typedef __attribute__((ext_vector_type(16))) __bf16 v16bf;
typedef __attribute__((ext_vector_type(8)))  float  v8f;
typedef __attribute__((ext_vector_type(4)))  unsigned int v4u;
typedef __attribute__((ext_vector_type(8)))  int    v8i;
typedef __attribute__((ext_vector_type(4)))  int    v4i;
using u16 = unsigned short;

union Frag { uint4 q[2]; v16bf v; };

__device__ __forceinline__ u16 f2bf(float f) {
    unsigned u = __builtin_bit_cast(unsigned, f);
    unsigned r = (u + 0x7FFFu + ((u >> 16) & 1u)) >> 16;
    return (u16)r;
}

__device__ __forceinline__ v8f wmma_bf16(v16bf a, v16bf b, v8f c) {
    return __builtin_amdgcn_wmma_f32_16x16x32_bf16(false, a, false, b, (short)0, c, false, false);
}

__device__ __forceinline__ unsigned lds_off(const void* p) {
    // generic "shared" addresses carry the LDS byte offset in the low 32 bits
    return (unsigned)(unsigned long long)p;
}

// ---------------------------------------------------------------------------
// Tensor Data Mover: 2D tile (tile_w x tile_h elements, 2-byte elements) from
// global (row stride = stride_elems) into LDS at lds_addr. D# per ISA 8.3/8.4.
// ---------------------------------------------------------------------------
__device__ __forceinline__ void tdm_load_2d(unsigned lds_addr, const void* gaddr,
                                            unsigned tile_w, unsigned tile_h,
                                            unsigned stride_elems) {
    unsigned long long ga = (unsigned long long)gaddr;
    v4u g0 = { 1u,                                   // count=1, valid user D#
               lds_addr,                             // LDS byte address
               (unsigned)ga,                         // global_addr[31:0]
               (unsigned)(ga >> 32) | 0x80000000u }; // global_addr[56:32] | type=2
    v8i g1 = { (int)0x00010000,                      // wg_mask=0, data_size=2B
               (int)((tile_w & 0xFFFFu) << 16),      // tensor_dim0[15:0]
               (int)((tile_w >> 16) | ((tile_h & 0xFFFFu) << 16)), // dim0 hi | dim1 lo
               (int)((tile_h >> 16) | (tile_w << 16)),             // dim1 hi | tile_dim0
               (int)tile_h,                          // tile_dim1 (tile_dim2=0)
               (int)stride_elems,                    // tensor_dim0_stride[31:0]
               0, 0 };
    v4i gz = { 0, 0, 0, 0 };
#if __clang_major__ >= 23
    v8i gz8 = { 0, 0, 0, 0, 0, 0, 0, 0 };
    __builtin_amdgcn_tensor_load_to_lds(g0, g1, gz, gz, gz8, 0);
#else
    __builtin_amdgcn_tensor_load_to_lds(g0, g1, gz, gz, 0);
#endif
}

// ---------------------------------------------------------------------------
// Problem constants: B=2, D=512, T=8, H=32, W=32; block (4,8,8); nb=64;
// thw=256; NH=8; DA=64; tokens = 16384
// ---------------------------------------------------------------------------
#define NTOK   16384
#define DMODEL 512

// ---------------------------------------------------------------------------
// Weight pack: f32 -> bf16, B^T [N][K] layout. Wqkv [1536][512],
// n = sec*512 + h*64 + e; value = w_sec[h][k][e]. 1/sqrt(64) folded into Q.
// ---------------------------------------------------------------------------
__global__ void pack_kernel(const float* __restrict__ wq, const float* __restrict__ wk,
                            const float* __restrict__ wv, const float* __restrict__ wp,
                            const float* __restrict__ w1, const float* __restrict__ w2,
                            u16* __restrict__ Wqkv, u16* __restrict__ Wp,
                            u16* __restrict__ W1t, u16* __restrict__ W2t) {
    const int QKV = 1536 * 512, S = 512 * 512;
    int idx = blockIdx.x * 256 + threadIdx.x;
    if (idx < QKV) {
        int n = idx >> 9, k = idx & 511;
        int sec = n >> 9, hi = n & 511, h = hi >> 6, e = hi & 63;
        const float* w = (sec == 0) ? wq : (sec == 1) ? wk : wv;
        float v = w[((size_t)h * 512 + k) * 64 + e];
        if (sec == 0) v *= 0.125f;
        Wqkv[idx] = f2bf(v);
    } else if (idx < QKV + S) {
        int i = idx - QKV; Wp[i] = f2bf(wp[i]);
    } else if (idx < QKV + 2 * S) {
        int i = idx - QKV - S; W1t[i] = f2bf(w1[i]);
    } else if (idx < QKV + 3 * S) {
        int i = idx - QKV - 2 * S; W2t[i] = f2bf(w2[i]);
    }
}

// ---------------------------------------------------------------------------
// Gather: x (B,C,T,H,W) -> resid f32 [nb][256][512]
// ---------------------------------------------------------------------------
__global__ void gather_kernel(const float* __restrict__ x, float* __restrict__ resid) {
    size_t idx = (size_t)blockIdx.x * 256 + threadIdx.x;
    int c = (int)(idx & 511);
    int tok = (int)(idx >> 9);
    int n = tok & 255, b = tok >> 8;
    int batch = b >> 5, ts = (b >> 4) & 1, hs = (b >> 2) & 3, ws = b & 3;
    int tt = n >> 6, hh = (n >> 3) & 7, ww = n & 7;
    size_t src = ((size_t)batch * 512 + c) * 8192 + (size_t)(ts * 4 + tt) * 1024
               + (size_t)(hs * 8 + hh) * 32 + (ws * 8 + ww);
    resid[idx] = x[src];
}

// ---------------------------------------------------------------------------
// LayerNorm over D=512: one wave per token, bf16 output
// ---------------------------------------------------------------------------
__global__ void ln_kernel(const float* __restrict__ in, const float* __restrict__ g,
                          const float* __restrict__ b, u16* __restrict__ out) {
    int token = blockIdx.x * 8 + (threadIdx.x >> 5);
    int lane = threadIdx.x & 31;
    const float* row = in + (size_t)token * DMODEL;
    float v[16];
    float s = 0.f;
#pragma unroll
    for (int i = 0; i < 16; ++i) { v[i] = row[lane + 32 * i]; s += v[i]; }
#pragma unroll
    for (int o = 16; o; o >>= 1) s += __shfl_xor(s, o);
    float mean = s * (1.0f / 512.0f);
    float vs = 0.f;
#pragma unroll
    for (int i = 0; i < 16; ++i) { float d = v[i] - mean; vs += d * d; }
#pragma unroll
    for (int o = 16; o; o >>= 1) vs += __shfl_xor(vs, o);
    float rstd = rsqrtf(vs * (1.0f / 512.0f) + 1e-5f);
    u16* orow = out + (size_t)token * DMODEL;
#pragma unroll
    for (int i = 0; i < 16; ++i) {
        int c = lane + 32 * i;
        orow[c] = f2bf((v[i] - mean) * rstd * g[c] + b[c]);
    }
}

// ---------------------------------------------------------------------------
// Tiled bf16 WMMA GEMM: C[M,N] = A[M,K] @ Bt[N,K]^T, 128x128 tile, 8 waves.
// MODE 0: bf16 store          MODE 1: f32 store + resid
// MODE 2: bf16 relu(C+bias)   MODE 3: f32 C+bias+resid scattered to (B,C,T,H,W)
// ---------------------------------------------------------------------------
template <int MODE>
__global__ void gemm_kernel(const u16* __restrict__ A, const u16* __restrict__ Bt,
                            int K, int N,
                            const float* __restrict__ bias, const float* __restrict__ resid,
                            u16* __restrict__ outb, float* __restrict__ outf) {
    __shared__ u16 sA[128 * 32];
    __shared__ u16 sB[128 * 32];
    int tid = threadIdx.x, lane = tid & 31, wave = tid >> 5;
    int bm = blockIdx.y * 128, bn = blockIdx.x * 128;
    int wm = wave >> 1, wn = wave & 1;      // 4x2 wave grid -> 32x64 per wave

    v8f acc[2][4];
#pragma unroll
    for (int i = 0; i < 2; ++i)
#pragma unroll
        for (int j = 0; j < 4; ++j) acc[i][j] = {};

    int row = tid >> 1, half = tid & 1;
    for (int k0 = 0; k0 < K; k0 += 32) {
        const uint4* ga = (const uint4*)&A[(size_t)(bm + row) * K + k0 + half * 16];
        uint4* da = (uint4*)&sA[row * 32 + half * 16];
        da[0] = ga[0]; da[1] = ga[1];
        const uint4* gb = (const uint4*)&Bt[(size_t)(bn + row) * K + k0 + half * 16];
        uint4* db = (uint4*)&sB[row * 32 + half * 16];
        db[0] = gb[0]; db[1] = gb[1];
        if (k0 + 32 < K) {  // L2-resident after first touch; hide first-touch latency
            __builtin_prefetch(&A[(size_t)(bm + row) * K + k0 + 32 + half * 16], 0, 0);
            __builtin_prefetch(&Bt[(size_t)(bn + row) * K + k0 + 32 + half * 16], 0, 0);
        }
        __syncthreads();

        Frag af[2], bf[4];
        int kh = (lane >> 4) * 8;         // A frag: split-K halves (ISA layout)
        int kb = (lane >> 4) * 8;         // B frag: split-K halves (ISA layout)
#pragma unroll
        for (int mt = 0; mt < 2; ++mt) {
            int arow = wm * 32 + mt * 16 + (lane & 15);
            af[mt].q[0] = *(const uint4*)&sA[arow * 32 + kh];
            af[mt].q[1] = *(const uint4*)&sA[arow * 32 + kh + 16];
        }
#pragma unroll
        for (int nt = 0; nt < 4; ++nt) {
            int nrow = wn * 64 + nt * 16 + (lane & 15);
            bf[nt].q[0] = *(const uint4*)&sB[nrow * 32 + kb];
            bf[nt].q[1] = *(const uint4*)&sB[nrow * 32 + kb + 16];
        }
#pragma unroll
        for (int mt = 0; mt < 2; ++mt)
#pragma unroll
            for (int nt = 0; nt < 4; ++nt)
                acc[mt][nt] = wmma_bf16(af[mt].v, bf[nt].v, acc[mt][nt]);
        __syncthreads();
    }

    // Epilogue: C layout -> row = vgpr + 8*(lane/16), col = lane%16
#pragma unroll
    for (int mt = 0; mt < 2; ++mt)
#pragma unroll
        for (int nt = 0; nt < 4; ++nt)
#pragma unroll
            for (int r = 0; r < 8; ++r) {
                int rowg = bm + wm * 32 + mt * 16 + r + 8 * (lane >> 4);
                int colg = bn + wn * 64 + nt * 16 + (lane & 15);
                float val = acc[mt][nt][r];
                size_t oi = (size_t)rowg * N + colg;
                if (MODE == 0) {
                    outb[oi] = f2bf(val);
                } else if (MODE == 1) {
                    outf[oi] = val + resid[oi];
                } else if (MODE == 2) {
                    val += bias[colg];
                    outb[oi] = f2bf(fmaxf(val, 0.f));
                } else {
                    val += bias[colg] + resid[oi];
                    int n = rowg & 255, b = rowg >> 8;
                    int batch = b >> 5, ts = (b >> 4) & 1, hs = (b >> 2) & 3, ws = b & 3;
                    int tt = n >> 6, hh = (n >> 3) & 7, ww = n & 7;
                    size_t dst = ((size_t)batch * 512 + colg) * 8192
                               + (size_t)(ts * 4 + tt) * 1024
                               + (size_t)(hs * 8 + hh) * 32 + (ws * 8 + ww);
                    outf[dst] = val;
                }
            }
}

// ---------------------------------------------------------------------------
// Fused attention: one workgroup per (block b, head h). Q/K/V staged by the
// Tensor Data Mover; S never leaves LDS (~288 KB of the 320 KB/WGP).
// ---------------------------------------------------------------------------
__global__ void attn_kernel(const u16* __restrict__ qkv,
                            const float* __restrict__ dt, const float* __restrict__ dh,
                            const float* __restrict__ dw, u16* __restrict__ attno) {
    extern __shared__ __align__(16) char smem[];
    u16*   sQ  = (u16*)smem;                  // [256][64]
    u16*   sK  = sQ + 256 * 64;               // [256][64]
    u16*   sVt = sK + 256 * 64;               // [64][256] (V transposed)
    float* sS  = (float*)(sVt + 64 * 256);    // 8 waves * [16][256]
    u16*   sP  = (u16*)(sS + 8 * 16 * 256);   // 8 waves * [16][256]; also V staging
    float* sDt = (float*)(sP + 8 * 16 * 256); // 7
    float* sDh = sDt + 7;                     // 15
    float* sDw = sDh + 15;                    // 15

    int b = blockIdx.x >> 3, h = blockIdx.x & 7;
    int tid = threadIdx.x, lane = tid & 31, wave = tid >> 5;

    if (tid < 7)       sDt[tid] = dt[h * 7 + tid];
    else if (tid < 22) sDh[tid - 7]  = dh[h * 15 + tid - 7];
    else if (tid < 37) sDw[tid - 22] = dw[h * 15 + tid - 22];

    // TDM: DMA Q,K into place and V into staging (sP, unused until softmax)
    if (tid < 32) {
        const u16* gq = qkv + (size_t)(b * 256) * 1536 + h * 64;
        tdm_load_2d(lds_off(sQ), gq,        64, 256, 1536);
        tdm_load_2d(lds_off(sK), gq + 512,  64, 256, 1536);
        tdm_load_2d(lds_off(sP), gq + 1024, 64, 256, 1536);
        __builtin_amdgcn_s_wait_tensorcnt(0);
    }
    __syncthreads();

    { // transpose staged V [256][64] -> sVt [64][256]
        int r = tid;
        union { uint4 q[8]; u16 s[64]; } vv;
        const uint4* pv = (const uint4*)&sP[r * 64];
#pragma unroll
        for (int i = 0; i < 8; ++i) vv.q[i] = pv[i];
#pragma unroll
        for (int c = 0; c < 64; ++c) sVt[c * 256 + r] = vv.s[c];
    }
    __syncthreads();

    // wc = col&7 is jn-invariant (tile stride 16 = 0 mod w); preload the 8
    // dw-bank values this lane ever needs into registers (wr == r exactly).
    float dwv[8];
    {
        int wc = lane & 7;
#pragma unroll
        for (int r = 0; r < 8; ++r) dwv[r] = sDw[r + 7 - wc];
    }

    const int rbase = wave * 32;
    float* Srow = sS + wave * 16 * 256;
    u16*   Pw   = sP + wave * 16 * 256;

    for (int mt = 0; mt < 2; ++mt) {
        int mrow = rbase + mt * 16;
        int rowOff = 8 * (lane >> 4);
        // tr/hr are constant per (wave, mtile, lane-half): rows mrow+rowOff+r,
        // r<8 never crosses an h- or t-boundary (w=8, t-block=64).
        int tr = (mrow + rowOff) >> 6;
        int hr = ((mrow + rowOff) >> 3) & 7;

        Frag aq0, aq1;
        {
            int arow = mrow + (lane & 15);
            int kh = (lane >> 4) * 8;
            aq0.q[0] = *(const uint4*)&sQ[arow * 64 + kh];
            aq0.q[1] = *(const uint4*)&sQ[arow * 64 + kh + 16];
            aq1.q[0] = *(const uint4*)&sQ[arow * 64 + 32 + kh];
            aq1.q[1] = *(const uint4*)&sQ[arow * 64 + 32 + kh + 16];
        }

        // S = Q K^T + axial bias, 16 N-tiles
#pragma unroll
        for (int jn = 0; jn < 16; ++jn) {
            Frag bk0, bk1;
            int ntok = jn * 16 + (lane & 15);
            int kb = (lane >> 4) * 8;
            bk0.q[0] = *(const uint4*)&sK[ntok * 64 + kb];
            bk0.q[1] = *(const uint4*)&sK[ntok * 64 + kb + 16];
            bk1.q[0] = *(const uint4*)&sK[ntok * 64 + 32 + kb];
            bk1.q[1] = *(const uint4*)&sK[ntok * 64 + 32 + kb + 16];
            v8f acc = {};
            acc = wmma_bf16(aq0.v, bk0.v, acc);
            acc = wmma_bf16(aq1.v, bk1.v, acc);

            int colTok = ntok;
            int tc = colTok >> 6, hc = (colTok >> 3) & 7;
            float dth = sDt[tr - tc + 3] + sDh[hr - hc + 7];  // hoisted: 2 loads/tile
#pragma unroll
            for (int r = 0; r < 8; ++r)
                Srow[(r + rowOff) * 256 + colTok] = acc[r] + dth + dwv[r];
        }
        asm volatile("s_wait_dscnt 0" ::: "memory");

        // row softmax: 2 lanes per row, 128 cols each
        {
            int srow = lane >> 1, seg = lane & 1;
            float* rp = Srow + srow * 256 + seg * 128;
            float mx = -1e30f;
#pragma unroll 4
            for (int c = 0; c < 128; ++c) mx = fmaxf(mx, rp[c]);
            mx = fmaxf(mx, __shfl_xor(mx, 1));
            float sum = 0.f;
#pragma unroll 4
            for (int c = 0; c < 128; ++c) { float e = __expf(rp[c] - mx); sum += e; rp[c] = e; }
            sum += __shfl_xor(sum, 1);
            float inv = 1.0f / sum;
            u16* pp = Pw + srow * 256 + seg * 128;
#pragma unroll 4
            for (int c = 0; c < 128; ++c) pp[c] = f2bf(rp[c] * inv);
        }
        asm volatile("s_wait_dscnt 0" ::: "memory");

        // O = P V  (K-dim = 256 sequence, 4 output N-tiles over DA=64)
#pragma unroll
        for (int nt = 0; nt < 4; ++nt) {
            v8f acc = {};
#pragma unroll
            for (int kc = 0; kc < 8; ++kc) {
                Frag ap, bv;
                int prow = lane & 15;
                int pk = kc * 32 + (lane >> 4) * 8;
                ap.q[0] = *(const uint4*)&Pw[prow * 256 + pk];
                ap.q[1] = *(const uint4*)&Pw[prow * 256 + pk + 16];
                int dac = nt * 16 + (lane & 15);
                int vk = kc * 32 + (lane >> 4) * 8;
                bv.q[0] = *(const uint4*)&sVt[dac * 256 + vk];
                bv.q[1] = *(const uint4*)&sVt[dac * 256 + vk + 16];
                acc = wmma_bf16(ap.v, bv.v, acc);
            }
#pragma unroll
            for (int r = 0; r < 8; ++r) {
                int rowTok = mrow + r + 8 * (lane >> 4);
                int col = h * 64 + nt * 16 + (lane & 15);
                attno[((size_t)(b * 256 + rowTok)) * 512 + col] = f2bf(acc[r]);
            }
        }
    }
}

// ---------------------------------------------------------------------------
extern "C" void kernel_launch(void* const* d_in, const int* in_sizes, int n_in,
                              void* d_out, int out_size, void* d_ws, size_t ws_size,
                              hipStream_t stream) {
    const float* x    = (const float*)d_in[0];
    const float* dtb  = (const float*)d_in[1];
    const float* dhb  = (const float*)d_in[2];
    const float* dwb  = (const float*)d_in[3];
    const float* ln1g = (const float*)d_in[4];
    const float* ln1b = (const float*)d_in[5];
    const float* wq   = (const float*)d_in[6];
    const float* wk   = (const float*)d_in[7];
    const float* wv   = (const float*)d_in[8];
    const float* wp   = (const float*)d_in[9];
    const float* ln2g = (const float*)d_in[10];
    const float* ln2b = (const float*)d_in[11];
    const float* w1   = (const float*)d_in[12];
    const float* b1   = (const float*)d_in[13];
    const float* w2   = (const float*)d_in[14];
    const float* b2   = (const float*)d_in[15];

    char* ws = (char*)d_ws;
    size_t off = 0;
    auto alloc = [&](size_t bytes) -> void* {
        void* p = ws + off;
        off = (off + bytes + 255) & ~(size_t)255;
        return p;
    };
    float* resid = (float*)alloc((size_t)NTOK * DMODEL * 4);
    float* out1  = (float*)alloc((size_t)NTOK * DMODEL * 4);
    u16* xn    = (u16*)alloc((size_t)NTOK * DMODEL * 2);
    u16* qkv   = (u16*)alloc((size_t)NTOK * 1536 * 2);
    u16* attno = (u16*)alloc((size_t)NTOK * DMODEL * 2);
    u16* yb    = (u16*)alloc((size_t)NTOK * DMODEL * 2);
    u16* ffb   = (u16*)alloc((size_t)NTOK * DMODEL * 2);
    u16* Wqkv  = (u16*)alloc((size_t)1536 * 512 * 2);
    u16* Wp    = (u16*)alloc((size_t)512 * 512 * 2);
    u16* W1t   = (u16*)alloc((size_t)512 * 512 * 2);
    u16* W2t   = (u16*)alloc((size_t)512 * 512 * 2);

    // 1) weight pack (bf16, B^T layout)
    pack_kernel<<<6144, 256, 0, stream>>>(wq, wk, wv, wp, w1, w2, Wqkv, Wp, W1t, W2t);
    // 2) block gather
    gather_kernel<<<32768, 256, 0, stream>>>(x, resid);
    // 3) LN1
    ln_kernel<<<2048, 256, 0, stream>>>(resid, ln1g, ln1b, xn);
    // 4) QKV GEMM: [16384,512] x [512,1536] -> bf16
    gemm_kernel<0><<<dim3(12, 128), 256, 0, stream>>>(xn, Wqkv, 512, 1536, nullptr, nullptr, qkv, nullptr);
    // 5) fused attention per (block, head), TDM-staged
    size_t attn_lds = (size_t)(256 * 64 * 2) * 2 + 64 * 256 * 2
                    + 8 * 16 * 256 * 4 + 8 * 16 * 256 * 2 + 256;
    attn_kernel<<<512, 256, attn_lds, stream>>>(qkv, dtb, dhb, dwb, attno);
    // 6) proj + residual -> out1 (f32)
    gemm_kernel<1><<<dim3(4, 128), 256, 0, stream>>>(attno, Wp, 512, 512, nullptr, resid, nullptr, out1);
    // 7) LN2
    ln_kernel<<<2048, 256, 0, stream>>>(out1, ln2g, ln2b, yb);
    // 8) FFN1: relu(y@w1^T + b1) -> bf16
    gemm_kernel<2><<<dim3(4, 128), 256, 0, stream>>>(yb, W1t, 512, 512, b1, nullptr, ffb, nullptr);
    // 9) FFN2 + b2 + residual, scattered back to (B,C,T,H,W)
    gemm_kernel<3><<<dim3(4, 128), 256, 0, stream>>>(ffb, W2t, 512, 512, b2, out1, nullptr, (float*)d_out);
}
